// Albertwithsoftmax_84920093376670
// MI455X (gfx1250) — matrix operand order repaired and checked by reading, not compiled, not run on hardware
//
#include <hip/hip_runtime.h>
#include <hip/hip_bf16.h>

typedef __attribute__((ext_vector_type(16))) _Float16 v16h;
typedef __attribute__((ext_vector_type(8)))  _Float16 v8h;
typedef __attribute__((ext_vector_type(8)))  float    v8f;

constexpr int B_   = 4;
constexpr int S_   = 512;
constexpr int D_   = 768;
constexpr int FF_  = 3072;
constexpr int V_   = 30000;
constexpr int H_   = 12;
constexpr int M_   = 64;
constexpr int DH_  = 64;
constexpr int VP_  = 30080;   // V padded to multiple of 128
constexpr int NLAYER_ = 12;

// ---------------------------------------------------------------------------
// WMMA fragment helpers: V_WMMA_F32_16X16X32_F16  (D = A[16x32] * B[32x16] + C)
// A layout (ISA 7.12.2, 16-bit A 16x32): lane<16 -> row=lane, halves 0..7 = K 0..7,
// halves 8..15 = K 16..23 ; lane>=16 -> row=lane-16, K 8..15 and K 24..31.
// B layout (32x16): lane<16 -> col N=lane, K 0..7 and K 16..23 ; lane>=16 -> col N=lane-16, K 8..15 and K 24..31.
// C/D layout: VGPR j -> M = m0 + j + 8*(lane>=16), N = n0 + (lane&15).
// ---------------------------------------------------------------------------
__device__ __forceinline__ v16h load_frag_a(const _Float16* __restrict__ base, int ld)
{
    const int lane = threadIdx.x & 31;
    const int r = lane & 15;
    const int c = (lane >> 4) << 3;            // 0 or 8
    const _Float16* p = base + (size_t)r * ld + c;
    v8h lo = *reinterpret_cast<const v8h*>(p);
    v8h hi = *reinterpret_cast<const v8h*>(p + 16);
    v16h out;
#pragma unroll
    for (int i = 0; i < 8; ++i) { out[i] = lo[i]; out[i + 8] = hi[i]; }
    return out;
}

__device__ __forceinline__ v16h load_frag_b(const _Float16* __restrict__ base, int ld)
{
    // base points at Wt tile origin [n0][k0], Wt is [N][K] row-major (i.e. B^T).
    const int lane = threadIdx.x & 31;
    const int n = lane & 15;
    const int c = (lane >> 4) << 3;            // K offset 0 or 8
    const _Float16* p = base + (size_t)n * ld + c;
    v8h lo = *reinterpret_cast<const v8h*>(p);
    v8h hi = *reinterpret_cast<const v8h*>(p + 16);
    v16h out;
#pragma unroll
    for (int i = 0; i < 8; ++i) { out[i] = lo[i]; out[i + 8] = hi[i]; }
    return out;
}

__device__ __forceinline__ v8f wmma_f16(v16h a, v16h b, v8f c)
{
    return __builtin_amdgcn_wmma_f32_16x16x32_f16(false, a, false, b, (short)0, c,
                                                  false, false);
}

__device__ __forceinline__ void store_frag(float* __restrict__ C32,
                                           _Float16* __restrict__ C16,
                                           int ldc, int m0, int n0, int nvalid,
                                           const float* __restrict__ bias,
                                           int relu, v8f acc)
{
    const int lane = threadIdx.x & 31;
    const int n = n0 + (lane & 15);
    if (n >= nvalid) return;
    const int mbase = m0 + ((lane >> 4) << 3);
    const float bv = bias ? bias[n] : 0.0f;
#pragma unroll
    for (int j = 0; j < 8; ++j) {
        float v = acc[j] + bv;
        if (relu) v = fmaxf(v, 0.0f);
        const size_t o = (size_t)(mbase + j) * ldc + n;
        if (C32) C32[o] = v;
        if (C16) C16[o] = (_Float16)v;
    }
}

// ---------------------------------------------------------------------------
// Generic GEMM:  C[M x N] = A_f16[M x K] * Wt_f16[N x K]^T (+ bias) (opt ReLU)
// Block = 256 threads = 8 waves arranged 4(M) x 2(N); wave tile 64x64
// (4 A-frags x 4 B-frags -> 16 WMMAs per 8 fragment loads); block tile 256x128.
// ---------------------------------------------------------------------------
__global__ __launch_bounds__(256) void gemm_f16_wmma(
    const _Float16* __restrict__ A, const _Float16* __restrict__ Wt,
    const float* __restrict__ bias, float* __restrict__ C32,
    _Float16* __restrict__ C16, int Mrows, int N, int K, int ldc, int relu)
{
    const int wave = threadIdx.x >> 5;
    const int m0 = blockIdx.y * 256 + (wave >> 1) * 64;
    const int n0 = blockIdx.x * 128 + (wave & 1) * 64;
    if (m0 >= Mrows) return;                    // wave-uniform guard

    const v8f vzero = {0.f, 0.f, 0.f, 0.f, 0.f, 0.f, 0.f, 0.f};
    v8f acc[4][4];
#pragma unroll
    for (int i = 0; i < 4; ++i)
#pragma unroll
        for (int j = 0; j < 4; ++j) acc[i][j] = vzero;

    for (int k = 0; k < K; k += 32) {
        v16h a[4];
#pragma unroll
        for (int i = 0; i < 4; ++i)
            a[i] = load_frag_a(A + (size_t)(m0 + 16 * i) * K + k, K);
#pragma unroll
        for (int j = 0; j < 4; ++j) {
            v16h b = load_frag_b(Wt + (size_t)(n0 + 16 * j) * K + k, K);
#pragma unroll
            for (int i = 0; i < 4; ++i)
                acc[i][j] = wmma_f16(a[i], b, acc[i][j]);
        }
    }
#pragma unroll
    for (int i = 0; i < 4; ++i)
#pragma unroll
        for (int j = 0; j < 4; ++j)
            store_frag(C32, C16, ldc, m0 + 16 * i, n0 + 16 * j, N, bias, relu,
                       acc[i][j]);
}

// ---------------------------------------------------------------------------
// Attention scores:  sc[b,h,q,k] = sum_dh Q[b,q,h,dh] * K[b,k,h,dh]
// (K rows are the columns of K^T -> B-frag reads K rows directly.)
// Block tile 256x128 over the 512x512 score matrix; wave tile 64x64.
// ---------------------------------------------------------------------------
__global__ __launch_bounds__(256) void attn_scores_wmma(
    const _Float16* __restrict__ q, const _Float16* __restrict__ k,
    float* __restrict__ sc)
{
    const int bh = blockIdx.z;
    const int b = bh / H_, h = bh - b * H_;
    const _Float16* qb = q + (size_t)b * S_ * D_ + h * DH_;
    const _Float16* kb = k + (size_t)b * S_ * D_ + h * DH_;
    float* scb = sc + (size_t)bh * S_ * S_;

    const int wave = threadIdx.x >> 5;
    const int m0 = blockIdx.y * 256 + (wave >> 1) * 64;
    const int n0 = blockIdx.x * 128 + (wave & 1) * 64;

    const v8f vzero = {0.f, 0.f, 0.f, 0.f, 0.f, 0.f, 0.f, 0.f};
    v8f acc[4][4];
#pragma unroll
    for (int i = 0; i < 4; ++i)
#pragma unroll
        for (int j = 0; j < 4; ++j) acc[i][j] = vzero;

#pragma unroll
    for (int kk = 0; kk < DH_; kk += 32) {
        v16h a[4];
#pragma unroll
        for (int i = 0; i < 4; ++i)
            a[i] = load_frag_a(qb + (size_t)(m0 + 16 * i) * D_ + kk, D_);
#pragma unroll
        for (int j = 0; j < 4; ++j) {
            v16h bf = load_frag_b(kb + (size_t)(n0 + 16 * j) * D_ + kk, D_);
#pragma unroll
            for (int i = 0; i < 4; ++i)
                acc[i][j] = wmma_f16(a[i], bf, acc[i][j]);
        }
    }
#pragma unroll
    for (int i = 0; i < 4; ++i)
#pragma unroll
        for (int j = 0; j < 4; ++j)
            store_frag(scb, nullptr, S_, m0 + 16 * i, n0 + 16 * j, S_, nullptr, 0,
                       acc[i][j]);
}

// ---------------------------------------------------------------------------
// o[b,q,h*DH+dh] = sum_k att[b,h,q,k] * V[b,k,h,dh]   (vt is [B,H,DH,S])
// One block per (b,h): 8 waves stacked along M -> block tile 512x64.
// ---------------------------------------------------------------------------
__global__ __launch_bounds__(256) void attn_av_wmma(
    const _Float16* __restrict__ att, const _Float16* __restrict__ vt,
    _Float16* __restrict__ o)
{
    const int bh = blockIdx.x;
    const int b = bh / H_, h = bh - b * H_;
    const _Float16* ab = att + (size_t)bh * S_ * S_;
    const _Float16* vb = vt + (size_t)bh * DH_ * S_;
    _Float16* ob = o + (size_t)b * S_ * D_ + h * DH_;

    const int wave = threadIdx.x >> 5;
    const int m0 = wave * 64;

    const v8f vzero = {0.f, 0.f, 0.f, 0.f, 0.f, 0.f, 0.f, 0.f};
    v8f acc[4][4];
#pragma unroll
    for (int i = 0; i < 4; ++i)
#pragma unroll
        for (int j = 0; j < 4; ++j) acc[i][j] = vzero;

    for (int kk = 0; kk < S_; kk += 32) {
        v16h a[4];
#pragma unroll
        for (int i = 0; i < 4; ++i)
            a[i] = load_frag_a(ab + (size_t)(m0 + 16 * i) * S_ + kk, S_);
#pragma unroll
        for (int j = 0; j < 4; ++j) {
            v16h bf = load_frag_b(vb + (size_t)(16 * j) * S_ + kk, S_);
#pragma unroll
            for (int i = 0; i < 4; ++i)
                acc[i][j] = wmma_f16(a[i], bf, acc[i][j]);
        }
    }
#pragma unroll
    for (int i = 0; i < 4; ++i)
#pragma unroll
        for (int j = 0; j < 4; ++j)
            store_frag(nullptr, ob, D_, m0 + 16 * i, 16 * j, DH_, nullptr, 0,
                       acc[i][j]);
}

// ---------------------------------------------------------------------------
// Softmax over one attention row (512), scale = 1/sqrt(64), f16 output.
// ---------------------------------------------------------------------------
__global__ __launch_bounds__(256) void softmax_rows(const float* __restrict__ sc,
                                                    _Float16* __restrict__ att)
{
    const float* r = sc + (size_t)blockIdx.x * S_;
    _Float16* o = att + (size_t)blockIdx.x * S_;
    __shared__ float red[256];
    const int t = threadIdx.x;
    const float v0 = r[t] * 0.125f;
    const float v1 = r[t + 256] * 0.125f;
    red[t] = fmaxf(v0, v1);
    __syncthreads();
    for (int s = 128; s > 0; s >>= 1) {
        if (t < s) red[t] = fmaxf(red[t], red[t + s]);
        __syncthreads();
    }
    const float m = red[0];
    __syncthreads();
    const float e0 = expf(v0 - m), e1 = expf(v1 - m);
    red[t] = e0 + e1;
    __syncthreads();
    for (int s = 128; s > 0; s >>= 1) {
        if (t < s) red[t] += red[t + s];
        __syncthreads();
    }
    const float inv = 1.0f / red[0];
    o[t] = (_Float16)(e0 * inv);
    o[t + 256] = (_Float16)(e1 * inv);
}

// ---------------------------------------------------------------------------
// x = LayerNorm(x + res) ; writes fp32 (residual chain) and f16 (GEMM input).
// ---------------------------------------------------------------------------
__global__ __launch_bounds__(256) void add_layernorm(const float* __restrict__ res,
                                                     float* __restrict__ x32,
                                                     _Float16* __restrict__ x16)
{
    float* xr = x32 + (size_t)blockIdx.x * D_;
    const float* rr = res + (size_t)blockIdx.x * D_;
    _Float16* hr = x16 + (size_t)blockIdx.x * D_;
    __shared__ float red[256];
    const int t = threadIdx.x;
    float v[3], s = 0.f;
#pragma unroll
    for (int j = 0; j < 3; ++j) { v[j] = xr[t + 256 * j] + rr[t + 256 * j]; s += v[j]; }
    red[t] = s;
    __syncthreads();
    for (int k = 128; k > 0; k >>= 1) { if (t < k) red[t] += red[t + k]; __syncthreads(); }
    const float mean = red[0] * (1.0f / D_);
    __syncthreads();
    s = 0.f;
#pragma unroll
    for (int j = 0; j < 3; ++j) { const float d = v[j] - mean; s += d * d; }
    red[t] = s;
    __syncthreads();
    for (int k = 128; k > 0; k >>= 1) { if (t < k) red[t] += red[t + k]; __syncthreads(); }
    const float inv = rsqrtf(red[0] * (1.0f / D_) + 1e-5f);
#pragma unroll
    for (int j = 0; j < 3; ++j) {
        const float y = (v[j] - mean) * inv;
        xr[t + 256 * j] = y;
        hr[t + 256 * j] = (_Float16)y;
    }
}

// ---------------------------------------------------------------------------
// Embedding: one-hot row -> token id gather + bias + positional encoding.
// ---------------------------------------------------------------------------
__global__ __launch_bounds__(256) void embed_kernel(const float* __restrict__ onehot,
                                                    const float* __restrict__ se_w,
                                                    const float* __restrict__ se_b,
                                                    float* __restrict__ x32,
                                                    _Float16* __restrict__ x16)
{
    const int bs = blockIdx.x;
    const int s = bs & (S_ - 1);
    const float* row = onehot + (size_t)bs * V_;
    __shared__ int tok;
    if (threadIdx.x == 0) tok = 0;
    __syncthreads();
    for (int i = threadIdx.x; i < V_; i += 256)
        if (row[i] > 0.5f) tok = i;
    __syncthreads();
    const int tk = tok;
    for (int d = threadIdx.x; d < D_; d += 256) {
        float pe = 0.0f;
        if (s > 0) {                         // row 0 stays raw pos/denom == 0
            const float denom = powf(10000.0f, 2.0f * (float)d / (float)D_);
            const float arg = (float)s / denom;
            pe = (d & 1) ? cosf(arg) : sinf(arg);
        }
        const float val = se_w[(size_t)tk * D_ + d] + se_b[d] + pe;
        x32[(size_t)bs * D_ + d] = val;
        x16[(size_t)bs * D_ + d] = (_Float16)val;
    }
}

// Wt[n][k] = W[k][n] (f32->f16); zero-fill padded rows n >= N.
__global__ __launch_bounds__(256) void w_transpose_f16(const float* __restrict__ W,
                                                       _Float16* __restrict__ Wt,
                                                       int K, int N, int Npad)
{
    const size_t i = (size_t)blockIdx.x * 256 + threadIdx.x;
    if (i >= (size_t)Npad * K) return;
    const int n = (int)(i / K);
    const int k = (int)(i - (size_t)n * K);
    Wt[i] = (n < N) ? (_Float16)W[(size_t)k * N + n] : (_Float16)0.0f;
}

// vt[b][h][dh][s] = v[b][s][h*DH+dh]
__global__ __launch_bounds__(256) void v_head_transpose(const _Float16* __restrict__ v,
                                                        _Float16* __restrict__ vt)
{
    const size_t i = (size_t)blockIdx.x * 256 + threadIdx.x;
    if (i >= (size_t)B_ * H_ * DH_ * S_) return;
    const int s = (int)(i & (S_ - 1));
    size_t r = i >> 9;
    const int dh = (int)(r & (DH_ - 1));
    r >>= 6;
    const int h = (int)(r % H_);
    const int b = (int)(r / H_);
    vt[i] = v[((size_t)b * S_ + s) * D_ + h * DH_ + dh];
}

// g[b*M+m][d] = x[b][idx[b][m]][d]  (f32 -> f16)
__global__ __launch_bounds__(256) void gather_rows(const float* __restrict__ x,
                                                   const int* __restrict__ idx,
                                                   _Float16* __restrict__ g)
{
    const size_t i = (size_t)blockIdx.x * 256 + threadIdx.x;
    if (i >= (size_t)B_ * M_ * D_) return;
    const int d = (int)(i % D_);
    const size_t r = i / D_;
    const int m = (int)(r & (M_ - 1));
    const int b = (int)(r >> 6);
    const int s = idx[b * M_ + m];
    g[i] = (_Float16)x[((size_t)b * S_ + s) * D_ + d];
}

// log_softmax over a 30000-wide row (logits have ld = VP_).
__global__ __launch_bounds__(256) void log_softmax_rows(const float* __restrict__ logits,
                                                        float* __restrict__ out)
{
    const float* L = logits + (size_t)blockIdx.x * VP_;
    float* O = out + (size_t)blockIdx.x * V_;
    __shared__ float red[256];
    const int t = threadIdx.x;
    float m = -3.0e38f;
    for (int i = t; i < V_; i += 256) m = fmaxf(m, L[i]);
    red[t] = m;
    __syncthreads();
    for (int s = 128; s > 0; s >>= 1) { if (t < s) red[t] = fmaxf(red[t], red[t + s]); __syncthreads(); }
    m = red[0];
    __syncthreads();
    float sum = 0.f;
    for (int i = t; i < V_; i += 256) sum += expf(L[i] - m);
    red[t] = sum;
    __syncthreads();
    for (int s = 128; s > 0; s >>= 1) { if (t < s) red[t] += red[t + s]; __syncthreads(); }
    const float lse = m + logf(red[0]);
    for (int i = t; i < V_; i += 256) O[i] = L[i] - lse;
}

// ---------------------------------------------------------------------------
extern "C" void kernel_launch(void* const* d_in, const int* in_sizes, int n_in,
                              void* d_out, int out_size, void* d_ws, size_t ws_size,
                              hipStream_t stream)
{
    (void)in_sizes; (void)n_in; (void)out_size; (void)ws_size;
    const float* onehot = (const float*)d_in[0];
    const float* se_w   = (const float*)d_in[1];
    const float* se_b   = (const float*)d_in[2];
    const float* wq = (const float*)d_in[3];  const float* bq = (const float*)d_in[4];
    const float* wk = (const float*)d_in[5];  const float* bk = (const float*)d_in[6];
    const float* wv = (const float*)d_in[7];  const float* bv = (const float*)d_in[8];
    const float* wo = (const float*)d_in[9];  const float* bo = (const float*)d_in[10];
    const float* w1 = (const float*)d_in[11]; const float* b1 = (const float*)d_in[12];
    const float* w2 = (const float*)d_in[13]; const float* b2 = (const float*)d_in[14];
    const float* we = (const float*)d_in[15]; const float* be = (const float*)d_in[16];
    const int*   idx = (const int*)d_in[17];
    float* out = (float*)d_out;

    char* wsp = (char*)d_ws;
    size_t off = 0;
    auto alloc = [&](size_t bytes) -> void* {
        void* p = wsp + off;
        off = (off + bytes + 255) & ~(size_t)255;
        return p;
    };

    _Float16* wqt  = (_Float16*)alloc((size_t)D_ * D_ * 2);
    _Float16* wkt  = (_Float16*)alloc((size_t)D_ * D_ * 2);
    _Float16* wvt  = (_Float16*)alloc((size_t)D_ * D_ * 2);
    _Float16* wot  = (_Float16*)alloc((size_t)D_ * D_ * 2);
    _Float16* w1t  = (_Float16*)alloc((size_t)FF_ * D_ * 2);
    _Float16* w2t  = (_Float16*)alloc((size_t)D_ * FF_ * 2);
    _Float16* wet  = (_Float16*)alloc((size_t)VP_ * D_ * 2);
    float*    x32  = (float*)   alloc((size_t)B_ * S_ * D_ * 4);
    _Float16* x16  = (_Float16*)alloc((size_t)B_ * S_ * D_ * 2);
    _Float16* q16  = (_Float16*)alloc((size_t)B_ * S_ * D_ * 2);
    _Float16* k16  = (_Float16*)alloc((size_t)B_ * S_ * D_ * 2);
    _Float16* v16b = (_Float16*)alloc((size_t)B_ * S_ * D_ * 2);
    _Float16* vt16 = (_Float16*)alloc((size_t)B_ * S_ * D_ * 2);
    float*    sc32 = (float*)   alloc((size_t)B_ * H_ * S_ * S_ * 4);
    _Float16* att16= (_Float16*)alloc((size_t)B_ * H_ * S_ * S_ * 2);
    _Float16* o16  = (_Float16*)alloc((size_t)B_ * S_ * D_ * 2);
    float*    ao32 = (float*)   alloc((size_t)B_ * S_ * D_ * 4);
    _Float16* ff16 = (_Float16*)alloc((size_t)B_ * S_ * FF_ * 2);
    _Float16* g16  = (_Float16*)alloc((size_t)B_ * M_ * D_ * 2);
    float*    lg32 = (float*)   alloc((size_t)B_ * M_ * VP_ * 4);

    auto cvt = [&](const float* W, _Float16* Wt, int K, int N, int Npad) {
        const size_t total = (size_t)Npad * K;
        w_transpose_f16<<<(unsigned)((total + 255) / 256), 256, 0, stream>>>(W, Wt, K, N, Npad);
    };
    cvt(wq, wqt, D_, D_, D_);
    cvt(wk, wkt, D_, D_, D_);
    cvt(wv, wvt, D_, D_, D_);
    cvt(wo, wot, D_, D_, D_);
    cvt(w1, w1t, D_, FF_, FF_);
    cvt(w2, w2t, FF_, D_, D_);
    cvt(we, wet, D_, V_, VP_);

    embed_kernel<<<B_ * S_, 256, 0, stream>>>(onehot, se_w, se_b, x32, x16);

    auto gemm = [&](const _Float16* A, const _Float16* Wt, const float* bias,
                    float* C32, _Float16* C16, int Mrows, int N, int K, int Npad,
                    int ldc, int relu) {
        dim3 grid(Npad / 128, (Mrows + 255) / 256);
        gemm_f16_wmma<<<grid, 256, 0, stream>>>(A, Wt, bias, C32, C16, Mrows, N, K,
                                                ldc, relu);
    };

    const int MR = B_ * S_;  // 2048 token rows
    for (int layer = 0; layer < NLAYER_; ++layer) {
        gemm(x16, wqt, bq, nullptr, q16, MR, D_, D_, D_, D_, 0);
        gemm(x16, wkt, bk, nullptr, k16, MR, D_, D_, D_, D_, 0);
        gemm(x16, wvt, bv, nullptr, v16b, MR, D_, D_, D_, D_, 0);
        {
            const size_t tot = (size_t)B_ * H_ * DH_ * S_;
            v_head_transpose<<<(unsigned)((tot + 255) / 256), 256, 0, stream>>>(v16b, vt16);
        }
        attn_scores_wmma<<<dim3(S_ / 128, S_ / 256, B_ * H_), 256, 0, stream>>>(q16, k16, sc32);
        softmax_rows<<<B_ * H_ * S_, 256, 0, stream>>>(sc32, att16);
        attn_av_wmma<<<B_ * H_, 256, 0, stream>>>(att16, vt16, o16);
        gemm(o16, wot, bo, ao32, nullptr, MR, D_, D_, D_, D_, 0);
        add_layernorm<<<MR, 256, 0, stream>>>(ao32, x32, x16);
        gemm(x16, w1t, b1, nullptr, ff16, MR, FF_, D_, FF_, FF_, 1);       // ReLU
        gemm(ff16, w2t, b2, x32, x16, MR, D_, FF_, D_, D_, 0);             // replaces x
    }

    {
        const size_t tot = (size_t)B_ * M_ * D_;
        gather_rows<<<(unsigned)((tot + 255) / 256), 256, 0, stream>>>(x32, idx, g16);
    }
    gemm(g16, wet, be, lg32, nullptr, B_ * M_, V_, D_, VP_, VP_, 0);
    log_softmax_rows<<<B_ * M_, 256, 0, stream>>>(lg32, out);
}
